// Qwen3VLAttention_17901423690406
// MI455X (gfx1250) — matrix operand order repaired and checked by reading, not compiled, not run on hardware
//
#include <hip/hip_runtime.h>
#include <hip/hip_bf16.h>

// ---------------------------------------------------------------------------
// Qwen3-VL attention block for gfx1250 (MI455X), bf16 WMMA path.
// H=32 q-heads, HKV=8 kv-heads, D=128, T=2048, HID=4096.
// ---------------------------------------------------------------------------

typedef __bf16 bf16_t;
typedef __attribute__((ext_vector_type(16))) __bf16 v16bf;
typedef __attribute__((ext_vector_type(8)))  __bf16 v8bf;
typedef __attribute__((ext_vector_type(8)))  float  v8f;

#define TT   2048
#define HID  4096
#define NH   32
#define NKV  8
#define DH   128

static __device__ __forceinline__ v8f wmma_bf16(v16bf a, v16bf b, v8f c) {
  // D = A(16x32) * B(32x16) + C, fp32 accumulate.
  return __builtin_amdgcn_wmma_f32_16x16x32_bf16(
      /*neg_a=*/false, a, /*neg_b=*/false, b,
      /*c_mod=*/(short)0, c, /*reuse_a=*/false, /*reuse_b=*/false);
}

// A-matrix fragment (16x32 bf16, ISA 7.12.2): lane = row (lane&15),
// element e -> K = (e<8?0:16) + (lane>>4)*8 + (e&7). Source row-major, ld elems.
static __device__ __forceinline__ v16bf load_fragA(const bf16_t* base, int ld, int lane) {
  const int row = lane & 15, kg = lane >> 4;
  const bf16_t* p = base + row * ld + kg * 8;
  v8bf lo = *(const v8bf*)(p);
  v8bf hi = *(const v8bf*)(p + 16);
  v16bf f;
#pragma unroll
  for (int i = 0; i < 8; ++i) { f[i] = lo[i]; f[i + 8] = hi[i]; }
  return f;
}

// B-matrix fragment (32x16 bf16): lane = col (lane&15), element e -> K = (e<8?0:16) + (lane>>4)*8 + (e&7).
// Source is "col-major": 32 contiguous K values per column, ld elems apart.
static __device__ __forceinline__ v16bf load_fragB(const bf16_t* base, int ld, int lane) {
  const int col = lane & 15, kg = lane >> 4;
  const bf16_t* p = base + col * ld + kg * 8;
  v8bf lo = *(const v8bf*)(p);
  v8bf hi = *(const v8bf*)(p + 16);
  v16bf f;
#pragma unroll
  for (int i = 0; i < 8; ++i) { f[i] = lo[i]; f[i + 8] = hi[i]; }
  return f;
}

static __device__ __forceinline__ v8f zero8() {
  v8f z = {0.f, 0.f, 0.f, 0.f, 0.f, 0.f, 0.f, 0.f};
  return z;
}

// LDS byte address of a __shared__ object (generic -> addrspace(3) -> u32).
static __device__ __forceinline__ unsigned lds_addr_of(const void* p) {
  return (unsigned)(unsigned long long)
         (__attribute__((address_space(3))) const char*)p;
}

// GLOBAL_LOAD_ASYNC_TO_LDS_B128 (ISA 15.18.3 op 98): per-lane 16B global -> LDS,
// tracked with ASYNCcnt. LDS dest byte address in a VGPR, 64-bit global addr in
// a VGPR pair.
static __device__ __forceinline__ void async_load_b128(unsigned lds_off, const void* gptr) {
  asm volatile("global_load_async_to_lds_b128 %0, %1, off"
               :: "v"(lds_off), "v"((unsigned long long)(__UINTPTR_TYPE__)gptr)
               : "memory");
}

static __device__ __forceinline__ void wait_asynccnt0() {
  asm volatile("s_wait_asynccnt 0x0" ::: "memory");
}

// ---------------------------------------------------------------------------
// fp32 -> bf16 convert (grid-stride)
// ---------------------------------------------------------------------------
__global__ void cvt_kernel(const float* __restrict__ src, bf16_t* __restrict__ dst, int n) {
  for (int i = blockIdx.x * blockDim.x + threadIdx.x; i < n; i += gridDim.x * blockDim.x)
    dst[i] = (bf16_t)src[i];
}

// ---------------------------------------------------------------------------
// C[M,N] fp32 = A[M,K] bf16 (row-major) * B[K,N] bf16 (row-major)
// 128x128 block tile, BK=32, 8 waves arranged 4(M) x 2(N): each wave 32x64,
// i.e. 2 A-frags x 4 B-frags -> 8 WMMAs per K-step.
// A tile is filled with GLOBAL_LOAD_ASYNC_TO_LDS_B128 (ASYNCcnt path);
// B tile is transposed into [col][k] through VGPRs.
// ---------------------------------------------------------------------------
__global__ __launch_bounds__(256) void gemm_bf16_kernel(
    const bf16_t* __restrict__ A, const bf16_t* __restrict__ B,
    float* __restrict__ C, int M, int N, int K) {
  __shared__ __align__(16) bf16_t As[128 * 32];  // [row][k]   8 KB
  __shared__ __align__(16) bf16_t Bs[128 * 32];  // [col][k]   8 KB

  const int tid  = threadIdx.x;
  const int lane = tid & 31;
  const int wave = tid >> 5;
  const int m0 = blockIdx.y * 128;
  const int n0 = blockIdx.x * 128;
  const int wm = (wave & 3) * 32;   // wave's M block (2 tiles)
  const int wn = (wave >> 2) * 64;  // wave's N block (4 tiles)

  const int arow = tid >> 1, aseg = (tid & 1) * 16;   // A: 128 rows x 2 segs of 16
  const int bk   = tid & 31, bc   = (tid >> 5) * 16;  // B: 32 k x 16-col groups

  v8f acc[2][4];
#pragma unroll
  for (int i = 0; i < 2; ++i)
#pragma unroll
    for (int j = 0; j < 4; ++j) acc[i][j] = zero8();

  const unsigned asl = lds_addr_of(&As[arow * 32 + aseg]);

  for (int k0 = 0; k0 < K; k0 += 32) {
    // A tile: two async 16B global->LDS transfers per thread.
    const bf16_t* ag = &A[(m0 + arow) * K + k0 + aseg];
    async_load_b128(asl,      ag);
    async_load_b128(asl + 16, ag + 8);

    // B tile: load 16 contiguous cols of one k-row, scatter transposed.
    v8bf bv0 = *(const v8bf*)&B[(k0 + bk) * N + n0 + bc];
    v8bf bv1 = *(const v8bf*)&B[(k0 + bk) * N + n0 + bc + 8];
#pragma unroll
    for (int i = 0; i < 8; ++i) {
      Bs[(bc + i)     * 32 + bk] = bv0[i];
      Bs[(bc + 8 + i) * 32 + bk] = bv1[i];
    }

    if (k0 + 32 < K) {  // keep next tiles warm in near caches
      __builtin_prefetch(&A[(m0 + arow) * K + k0 + 32 + aseg], 0, 3);
      __builtin_prefetch(&B[(k0 + 32 + bk) * N + n0 + bc], 0, 3);
    }

    wait_asynccnt0();   // this wave's async LDS writes landed
    __syncthreads();    // all waves' fills (ds + async) visible

    v16bf af[2], bfg[4];
#pragma unroll
    for (int i = 0; i < 2; ++i) af[i]  = load_fragA(&As[(wm + i * 16) * 32], 32, lane);
#pragma unroll
    for (int j = 0; j < 4; ++j) bfg[j] = load_fragB(&Bs[(wn + j * 16) * 32], 32, lane);
#pragma unroll
    for (int i = 0; i < 2; ++i)
#pragma unroll
      for (int j = 0; j < 4; ++j) acc[i][j] = wmma_bf16(af[i], bfg[j], acc[i][j]);
    __syncthreads();
  }

  const int kg = lane >> 4, cn = lane & 15;
#pragma unroll
  for (int i = 0; i < 2; ++i)
#pragma unroll
    for (int j = 0; j < 4; ++j)
#pragma unroll
      for (int r = 0; r < 8; ++r) {
        const int row = m0 + wm + i * 16 + r + 8 * kg;
        C[row * N + n0 + wn + j * 16 + cn] = acc[i][j][r];
      }
}

// ---------------------------------------------------------------------------
// Per-(t, head) RMSNorm + mRoPE. s<32: q-head, s<40: k-head, else v-head.
// Q -> Qb[h][t][128], K -> Kb[kvh][t][128], V -> VbT[kvh][d][t] (transposed).
// ---------------------------------------------------------------------------
__global__ __launch_bounds__(128) void norm_rope_kernel(
    const float* __restrict__ Qf, const float* __restrict__ Kf, const float* __restrict__ Vf,
    const float* __restrict__ cosT, const float* __restrict__ sinT,
    const float* __restrict__ qw, const float* __restrict__ kw,
    bf16_t* __restrict__ Qb, bf16_t* __restrict__ Kb, bf16_t* __restrict__ VbT) {
  const int d = threadIdx.x;
  const int t = blockIdx.x / (NH + 2 * NKV);
  const int s = blockIdx.x % (NH + 2 * NKV);

  __shared__ float red[DH];
  __shared__ float ybuf[DH];

  if (s < NH + NKV) {
    const bool isq = s < NH;
    const int  h   = isq ? s : (s - NH);
    const float v = isq ? Qf[(t * NH + h) * DH + d] : Kf[(t * NKV + h) * DH + d];
    red[d] = v * v;
    __syncthreads();
#pragma unroll
    for (int off = 64; off > 0; off >>= 1) {
      if (d < off) red[d] += red[d + off];
      __syncthreads();
    }
    const float inv = rsqrtf(red[0] * (1.f / DH) + 1e-6f);
    const float y = v * inv * (isq ? qw[d] : kw[d]);
    ybuf[d] = y;
    __syncthreads();
    // mRoPE: sections (24,20,20) pick table axis; duplicated halves.
    const int j = d & 63;
    const int axis = (j < 24) ? 0 : ((j < 44) ? 1 : 2);
    const float c  = cosT[(axis * TT + t) * DH + j];
    const float sn = sinT[(axis * TT + t) * DH + j];
    const float rot = (d < 64) ? -ybuf[d + 64] : ybuf[d - 64];
    const float o = y * c + rot * sn;
    if (isq) Qb[(h * TT + t) * DH + d] = (bf16_t)o;
    else     Kb[(h * TT + t) * DH + d] = (bf16_t)o;
  } else {
    const int h = s - (NH + NKV);
    VbT[(h * DH + d) * TT + t] = (bf16_t)Vf[(t * NKV + h) * DH + d];
  }
}

// ---------------------------------------------------------------------------
// Flash attention: one wave per (head, 16-row q tile). 32-key KV blocks,
// causal + padding mask, online softmax, WMMA QK^T and PV.
// Output Ob[t][h*128 + d] bf16.
// ---------------------------------------------------------------------------
__global__ __launch_bounds__(32) void flash_kernel(
    const bf16_t* __restrict__ Qb, const bf16_t* __restrict__ Kb,
    const bf16_t* __restrict__ VbT, const float* __restrict__ amask,
    bf16_t* __restrict__ Ob) {
  const int lane = threadIdx.x & 31;
  const int kg = lane >> 4, ln = lane & 15;
  const int h  = blockIdx.x >> 7;       // TT/16 = 128 q tiles per head
  const int qt = blockIdx.x & 127;
  const int q0 = qt * 16;
  const int kvh = h >> 2;               // GQA: 4 q-heads per kv head

  __shared__ __align__(16) bf16_t Ps[16 * 32];

  v16bf qf[4];
#pragma unroll
  for (int c = 0; c < 4; ++c)
    qf[c] = load_fragA(Qb + (h * TT + q0) * DH + c * 32, DH, lane);

  v8f o[8];
#pragma unroll
  for (int i = 0; i < 8; ++i) o[i] = zero8();
  float mstat[8], lstat[8];
#pragma unroll
  for (int r = 0; r < 8; ++r) { mstat[r] = -3.0e38f; lstat[r] = 0.f; }

  const float scale = 0.0883883476483184f;  // 128^-0.5
  const int nkv = (q0 + 15) / 32 + 1;

  for (int jb = 0; jb < nkv; ++jb) {
    const int kv0 = jb * 32;
    v8f s0 = zero8(), s1 = zero8();
#pragma unroll
    for (int c = 0; c < 4; ++c) {
      v16bf k0f = load_fragB(Kb + (kvh * TT + kv0)      * DH + c * 32, DH, lane);
      v16bf k1f = load_fragB(Kb + (kvh * TT + kv0 + 16) * DH + c * 32, DH, lane);
      s0 = wmma_bf16(qf[c], k0f, s0);
      s1 = wmma_bf16(qf[c], k1f, s1);
    }
    const float pm0 = (1.f - amask[kv0 + ln])      * -1e9f;
    const float pm1 = (1.f - amask[kv0 + 16 + ln]) * -1e9f;
#pragma unroll
    for (int r = 0; r < 8; ++r) {
      const int row = q0 + r + 8 * kg;
      float a0 = s0[r] * scale + pm0 + (((kv0 + ln)      > row) ? -1e9f : 0.f);
      float a1 = s1[r] * scale + pm1 + (((kv0 + 16 + ln) > row) ? -1e9f : 0.f);
      float rm = fmaxf(a0, a1);
      rm = fmaxf(rm, __shfl_xor(rm, 1, 32));
      rm = fmaxf(rm, __shfl_xor(rm, 2, 32));
      rm = fmaxf(rm, __shfl_xor(rm, 4, 32));
      rm = fmaxf(rm, __shfl_xor(rm, 8, 32));
      const float mnew = fmaxf(mstat[r], rm);
      const float p0 = __expf(a0 - mnew);
      const float p1 = __expf(a1 - mnew);
      float rs = p0 + p1;
      rs += __shfl_xor(rs, 1, 32);
      rs += __shfl_xor(rs, 2, 32);
      rs += __shfl_xor(rs, 4, 32);
      rs += __shfl_xor(rs, 8, 32);
      const float alpha = __expf(mstat[r] - mnew);
      lstat[r] = lstat[r] * alpha + rs;
      mstat[r] = mnew;
#pragma unroll
      for (int c = 0; c < 8; ++c) o[c][r] *= alpha;
      Ps[(r + 8 * kg) * 32 + ln]      = (bf16_t)p0;   // C-layout -> LDS
      Ps[(r + 8 * kg) * 32 + 16 + ln] = (bf16_t)p1;
    }
    __syncthreads();
    v16bf pf = load_fragA(Ps, 32, lane);              // LDS -> A-fragment layout
#pragma unroll
    for (int c = 0; c < 8; ++c) {
      v16bf vf = load_fragB(VbT + (kvh * DH + c * 16) * TT + kv0, TT, lane);
      o[c] = wmma_bf16(pf, vf, o[c]);
    }
    __syncthreads();
  }

#pragma unroll
  for (int r = 0; r < 8; ++r) {
    const float inv = 1.f / lstat[r];
    const int row = q0 + r + 8 * kg;
#pragma unroll
    for (int c = 0; c < 8; ++c)
      Ob[row * (NH * DH) + h * DH + c * 16 + ln] = (bf16_t)(o[c][r] * inv);
  }
}

// ---------------------------------------------------------------------------
// Host-side launch
// ---------------------------------------------------------------------------
extern "C" void kernel_launch(void* const* d_in, const int* in_sizes, int n_in,
                              void* d_out, int out_size, void* d_ws, size_t ws_size,
                              hipStream_t stream) {
  (void)in_sizes; (void)n_in; (void)out_size; (void)ws_size;
  const float* x     = (const float*)d_in[0];
  const float* cosT  = (const float*)d_in[1];
  const float* sinT  = (const float*)d_in[2];
  const float* amask = (const float*)d_in[3];
  const float* Wq    = (const float*)d_in[4];
  const float* Wk    = (const float*)d_in[5];
  const float* Wv    = (const float*)d_in[6];
  const float* Wo    = (const float*)d_in[7];
  const float* qw    = (const float*)d_in[8];
  const float* kw    = (const float*)d_in[9];
  float* out = (float*)d_out;

  char* ws = (char*)d_ws;
  size_t off = 0;
  auto alloc = [&](size_t bytes) -> void* {
    void* p = ws + off;
    off += (bytes + 255) & ~(size_t)255;
    return p;
  };
  bf16_t* xb  = (bf16_t*)alloc((size_t)TT * HID * 2);
  bf16_t* WqB = (bf16_t*)alloc((size_t)HID * (NH * DH) * 2);
  bf16_t* WkB = (bf16_t*)alloc((size_t)HID * (NKV * DH) * 2);
  bf16_t* WvB = (bf16_t*)alloc((size_t)HID * (NKV * DH) * 2);
  bf16_t* WoB = (bf16_t*)alloc((size_t)(NH * DH) * HID * 2);
  float*  Qf  = (float*) alloc((size_t)TT * (NH * DH) * 4);
  float*  Kf  = (float*) alloc((size_t)TT * (NKV * DH) * 4);
  float*  Vf  = (float*) alloc((size_t)TT * (NKV * DH) * 4);
  bf16_t* Qb  = (bf16_t*)alloc((size_t)NH * TT * DH * 2);
  bf16_t* Kb  = (bf16_t*)alloc((size_t)NKV * TT * DH * 2);
  bf16_t* VbT = (bf16_t*)alloc((size_t)NKV * DH * TT * 2);
  bf16_t* Ob  = (bf16_t*)alloc((size_t)TT * (NH * DH) * 2);

  // 1) fp32 -> bf16 conversions
  cvt_kernel<<<4096, 256, 0, stream>>>(x,  xb,  TT * HID);
  cvt_kernel<<<4096, 256, 0, stream>>>(Wq, WqB, HID * NH * DH);
  cvt_kernel<<<4096, 256, 0, stream>>>(Wk, WkB, HID * NKV * DH);
  cvt_kernel<<<4096, 256, 0, stream>>>(Wv, WvB, HID * NKV * DH);
  cvt_kernel<<<4096, 256, 0, stream>>>(Wo, WoB, NH * DH * HID);

  // 2) QKV projections (fp32 accumulate)
  {
    dim3 gq((NH * DH) / 128, TT / 128);
    gemm_bf16_kernel<<<gq, 256, 0, stream>>>(xb, WqB, Qf, TT, NH * DH, HID);
    dim3 gk((NKV * DH) / 128, TT / 128);
    gemm_bf16_kernel<<<gk, 256, 0, stream>>>(xb, WkB, Kf, TT, NKV * DH, HID);
    gemm_bf16_kernel<<<gk, 256, 0, stream>>>(xb, WvB, Vf, TT, NKV * DH, HID);
  }

  // 3) RMSNorm + mRoPE + layout change to attention-friendly bf16
  norm_rope_kernel<<<TT * (NH + 2 * NKV), DH, 0, stream>>>(
      Qf, Kf, Vf, cosT, sinT, qw, kw, Qb, Kb, VbT);

  // 4) causal GQA flash attention
  flash_kernel<<<NH * (TT / 16), 32, 0, stream>>>(Qb, Kb, VbT, amask, Ob);

  // 5) output projection -> fp32 d_out
  {
    dim3 go(HID / 128, TT / 128);
    gemm_bf16_kernel<<<go, 256, 0, stream>>>(Ob, WoB, out, TT, HID, NH * DH);
  }
}
